// CNN_VAE_split_MI_11304353923318
// MI455X (gfx1250) — matrix operand order repaired and checked by reading, not compiled, not run on hardware
//
#include <hip/hip_runtime.h>
#include <hip/hip_bf16.h>

typedef __attribute__((ext_vector_type(16))) _Float16 v16h;
typedef __attribute__((ext_vector_type(8)))  _Float16 v8h;
typedef __attribute__((ext_vector_type(4)))  _Float16 v4h;
typedef __attribute__((ext_vector_type(8)))  float    v8f;
typedef __attribute__((ext_vector_type(4)))  unsigned u32x4;
typedef __attribute__((ext_vector_type(8)))  int      i32x8;
typedef __attribute__((ext_vector_type(4)))  int      i32x4;

// ---- Tensor Data Mover availability / arity selection -----------------------
#if defined(__has_builtin)
#  if __has_builtin(__builtin_amdgcn_tensor_load_to_lds)
#    if defined(__has_include)
#      if __has_include(<hip/amd_detail/amd_gfx1250_TDM.h>)
#        define USE_TDM 2   /* amdgpu-toolchain: 6-arg builtin */
#      else
#        define USE_TDM 1   /* ROCm 7.2: 5-arg builtin */
#      endif
#    else
#      define USE_TDM 1
#    endif
#  else
#    define USE_TDM 0
#  endif
#else
#  define USE_TDM 0
#endif

__device__ __forceinline__ float lrelu(float v) { return v >= 0.f ? v : 0.2f * v; }

#if USE_TDM
// 2-D TDM load: tileH x tileW f16 tile (row-major, stride strideElems) from a
// global f16 tensor into LDS at byte offset ldsOff. D# packed per ISA 8.3/8.4.
__device__ __forceinline__ void tdm_load_tile_f16(unsigned ldsOff, const _Float16* gptr,
                                                  int tileW, int tileH,
                                                  int strideElems, int tensorW, int tensorH)
{
    const unsigned long long ga = (unsigned long long)(size_t)gptr;
    u32x4 g0;
    g0[0] = 1u;                                          // count=1, user descriptor
    g0[1] = ldsOff;                                      // lds_addr
    g0[2] = (unsigned)(ga & 0xffffffffu);                // global_addr[31:0]
    g0[3] = (unsigned)((ga >> 32) & 0x01ffffffu) | 0x80000000u;  // addr[56:32] | type=2
    i32x8 g1;
    g1[0] = (1 << 16);                                   // data_size=1 (2 bytes)
    g1[1] = (tensorW & 0xffff) << 16;                    // tensor_dim0[15:0]
    g1[2] = ((tensorW >> 16) & 0xffff) | ((tensorH & 0xffff) << 16);
    g1[3] = ((tensorH >> 16) & 0xffff) | ((tileW & 0xffff) << 16);   // tile_dim0
    g1[4] = (tileH & 0xffff);                            // tile_dim1 (tile_dim2=0)
    g1[5] = strideElems;                                 // tensor_dim0_stride[31:0]
    g1[6] = 0;
    g1[7] = 0;
    const i32x4 z4 = {0, 0, 0, 0};
#if USE_TDM == 2
    const i32x8 z8 = {0, 0, 0, 0, 0, 0, 0, 0};
    __builtin_amdgcn_tensor_load_to_lds(g0, g1, z4, z4, z8, 0);
#else
    __builtin_amdgcn_tensor_load_to_lds(g0, g1, z4, z4, 0);
#endif
}
#endif

// ---------------------------------------------------------------------------
// Weight packs.
// Encoder: fp32 (Cout x Ktot) -> f16 (Cout x Kpad), zero K-padding.
// ---------------------------------------------------------------------------
__global__ void pack_w_f16_kernel(const float* __restrict__ w, _Float16* __restrict__ wf,
                                  int Cout, int Ktot, int Kpad)
{
    const int i = blockIdx.x * blockDim.x + threadIdx.x;
    if (i >= Cout * Kpad) return;
    const int m = i / Kpad, k = i % Kpad;
    wf[i] = (_Float16)((k < Ktot) ? w[m * Ktot + k] : 0.f);
}

// Decoder (convT): w (Cin, Cout, 4, 4) -> f16 wt (Mpad x Cin*16), spatially
// flipped, M zero-padded: wt[o][ci*16+kh*4+kw] = w[ci][o][3-kh][3-kw].
__global__ void pack_wT_f16_kernel(const float* __restrict__ w, _Float16* __restrict__ wf,
                                   int Cin, int Cout, int Mpad)
{
    const int Kpad = Cin * 16;
    const int i = blockIdx.x * blockDim.x + threadIdx.x;
    if (i >= Mpad * Kpad) return;
    const int o = i / Kpad, k = i % Kpad;
    const int ci = k >> 4, r = k & 15;
    const int kh = r >> 2, kw = r & 3;
    float v = 0.f;
    if (o < Cout) v = w[((size_t)(ci * Cout + o)) * 16 + (3 - kh) * 4 + (3 - kw)];
    wf[i] = (_Float16)v;
}

// ---------------------------------------------------------------------------
// Unified implicit-GEMM conv / convT on v_wmma_f32_16x16x32_f16.
// Workgroup = 4 waves: one 16(M) x 64(N) tile, double-buffered LDS stages.
// A tiles via TDM (DMA overlapped with WMMA), B via branchless gather with
// K-loop-invariant tap offsets. All spatial sizes are powers of two, so
// indexing is shift/mask only. mode: 0=conv s2p1k4, 1=convT s2p1k4.
// epilogue: 0=raw, 1=lrelu, 2=sigmoid.
// ---------------------------------------------------------------------------
__global__ __launch_bounds__(128)
void gemm_conv_wmma_kernel(const float* __restrict__ x, const _Float16* __restrict__ wpk,
                           float* __restrict__ out, int Cin, int Cout, int Mrows,
                           int IH, int IW, int ihwSh, int owSh, int ohwSh,
                           int Kpad, int mode, int epilogue)
{
    __shared__ __attribute__((aligned(32))) _Float16 As[2][16][32];  // row-major (m,k)
    __shared__ __attribute__((aligned(32))) _Float16 Bs[2][64][32];  // col-major (n,k)

    const int tid    = threadIdx.x;
    const int lane   = tid & 31;
    const int wave   = tid >> 5;
    const int half   = lane >> 4;
    const int lane16 = lane & 15;

    const int mBase   = blockIdx.x * 16;
    const int nBase   = blockIdx.y * 64;
    const int ohwMask = (1 << ohwSh) - 1;
    const int owMask  = (1 << owSh) - 1;

    // Gather decode: this thread owns column nrel, k-half hb.
    const int nrel  = tid >> 1;
    const int hb    = tid & 1;
    const int nColF = nBase + nrel;
    const int fb    = nColF >> ohwSh;
    const int fp    = nColF & ohwMask;
    const int fbase = fb * Cin;

    // K-loop-invariant tap offsets + validity (16 taps, r == t always).
    int  off[16];
    bool val[16];
    if (mode == 0) {
        const int iy0 = ((fp >> owSh) << 1) - 1;     // stride 2, pad 1
        const int ix0 = ((fp & owMask) << 1) - 1;
#pragma unroll
        for (int t = 0; t < 16; ++t) {
            const int iy = iy0 + (t >> 2), ix = ix0 + (t & 3);
            const bool v = (iy >= 0) & (iy < IH) & (ix >= 0) & (ix < IW);
            val[t] = v;
            off[t] = v ? (iy * IW + ix) : 0;
        }
    } else {
        const int oy = fp >> owSh, ox = fp & owMask;
#pragma unroll
        for (int t = 0; t < 16; ++t) {
            const int dy = oy + (t >> 2) - 2, dx = ox + (t & 3) - 2;
            const bool vy = (dy >= 0) & ((dy & 1) == 0) & ((dy >> 1) < IH);
            const bool vx = (dx >= 0) & ((dx & 1) == 0) & ((dx >> 1) < IW);
            const bool v = vy & vx;
            val[t] = v;
            off[t] = v ? ((dy >> 1) * IW + (dx >> 1)) : 0;
        }
    }

    // Output column for this lane (per-wave n-subtile).
    const int nCol = nBase + wave * 16 + lane16;
    const int ob   = nCol >> ohwSh;
    const int op   = nCol & ohwMask;

#if USE_TDM
    const unsigned asOff0 = (unsigned)(size_t)(&As[0][0][0]);
    const unsigned asOff1 = (unsigned)(size_t)(&As[1][0][0]);
#endif

    auto fillStage = [&](int sel, int kks) {
#if USE_TDM
        if (tid < 32)
            tdm_load_tile_f16(sel ? asOff1 : asOff0,
                              wpk + (size_t)mBase * Kpad + kks,
                              32, 16, Kpad, Kpad, Mrows);
#else
        {   // cooperative A-tile copy: 128 threads x 4 contiguous halves
            const int e = tid * 4;
            const int m = e >> 5, kq = e & 31;
            *(v4h*)&As[sel][m][kq] = *(const v4h*)&wpk[(size_t)(mBase + m) * Kpad + kks + kq];
        }
#endif
        // This thread's 16 B elements share one input channel at this K-step.
        const int  ci = (kks >> 4) + hb;
        const bool cv = ci < Cin;
        const float* xb = x + ((size_t)(fbase + (cv ? ci : 0)) << ihwSh);
        v16h bvec;
#pragma unroll
        for (int t = 0; t < 16; ++t) {
            const float f = xb[off[t]];                 // unconditional, pipelined
            bvec[t] = (_Float16)((cv & val[t]) ? f : 0.f);
        }
        *(v16h*)&Bs[sel][nrel][hb * 16] = bvec;
    };

    fillStage(0, 0);
#if USE_TDM
    if (tid < 32) __builtin_amdgcn_s_wait_tensorcnt(0);
#endif
    __syncthreads();

    v8f acc = {};
    for (int kk = 0; kk < Kpad; kk += 32) {
        const int cur = (kk >> 5) & 1;
        const bool more = (kk + 32) < Kpad;
        if (more) fillStage(cur ^ 1, kk + 32);          // overlaps WMMA below

        const v8h a0 = *(const v8h*)&As[cur][lane16][half * 8];
        const v8h a1 = *(const v8h*)&As[cur][lane16][16 + half * 8];
        const v8h b0 = *(const v8h*)&Bs[cur][wave * 16 + lane16][half * 8];
        const v8h b1 = *(const v8h*)&Bs[cur][wave * 16 + lane16][16 + half * 8];
        const v16h af = __builtin_shufflevector(a0, a1, 0,1,2,3,4,5,6,7,8,9,10,11,12,13,14,15);
        const v16h bf = __builtin_shufflevector(b0, b1, 0,1,2,3,4,5,6,7,8,9,10,11,12,13,14,15);
        acc = __builtin_amdgcn_wmma_f32_16x16x32_f16(false, af, false, bf,
                                                     (short)0, acc, false, false);
#if USE_TDM
        if (more && tid < 32) __builtin_amdgcn_s_wait_tensorcnt(0);
#endif
        __syncthreads();
    }

    // D layout: lane holds N=lane16, rows M = r + 8*half.
#pragma unroll
    for (int r = 0; r < 8; ++r) {
        const int m = mBase + r + 8 * half;
        if (m < Cout) {
            float v = acc[r];
            if (epilogue == 1) v = lrelu(v);
            else if (epilogue == 2) v = 1.f / (1.f + expf(-v));
            out[(((size_t)(ob * Cout + m)) << ohwSh) + op] = v;
        }
    }
}

// ---------------------------------------------------------------------------
// BatchNorm stats + fused normalize/LeakyReLU (HW is a power of two)
// ---------------------------------------------------------------------------
__global__ __launch_bounds__(256)
void bn_stats_kernel(const float* __restrict__ x, float* __restrict__ meanOut,
                     float* __restrict__ rstdOut, int Bn, int C, int hwSh)
{
    __shared__ float s1[256], s2[256];
    const int c = blockIdx.x;
    const int hwMask = (1 << hwSh) - 1;
    const int N = Bn << hwSh;
    float a = 0.f, b = 0.f;
    for (int i = threadIdx.x; i < N; i += 256) {
        const int bb = i >> hwSh, p = i & hwMask;
        const float v = x[(((size_t)(bb * C + c)) << hwSh) + p];
        a += v; b += v * v;
    }
    s1[threadIdx.x] = a; s2[threadIdx.x] = b;
    __syncthreads();
    for (int s = 128; s > 0; s >>= 1) {
        if (threadIdx.x < s) { s1[threadIdx.x] += s1[threadIdx.x + s];
                               s2[threadIdx.x] += s2[threadIdx.x + s]; }
        __syncthreads();
    }
    if (threadIdx.x == 0) {
        const float m   = s1[0] / (float)N;
        const float var = s2[0] / (float)N - m * m;
        meanOut[c] = m;
        rstdOut[c] = rsqrtf(var + 1e-5f);
    }
}

// One block per (b,c): channel index resolved once, threads sweep HW.
__global__ __launch_bounds__(256)
void bn_lrelu_kernel(float* __restrict__ x, const float* __restrict__ mean,
                     const float* __restrict__ rstd, int C, int hwSh)
{
    const int blk = blockIdx.x;             // b*C + c
    const int c   = blk % C;
    const int HW  = 1 << hwSh;
    const float mu = mean[c], rs = rstd[c];
    float* xb = x + ((size_t)blk << hwSh);
    for (int p = threadIdx.x; p < HW; p += 256)
        xb[p] = lrelu((xb[p] - mu) * rs);
}

// ---------------------------------------------------------------------------
// Mutual information head
// ---------------------------------------------------------------------------
__global__ void mi_k_kernel(const float* __restrict__ h5, float* __restrict__ Kb)
{
    const int i = blockIdx.x * blockDim.x + threadIdx.x;
    if (i >= 8 * 96 * 16 * 64) return;
    const int bin = i & 63;
    const int idx = i >> 6;                 // (b*96+c)*16+n
    const float val = h5[idx] * 255.f;
    const float bc  = (float)bin * (255.f / 63.f);
    const float t   = (val - bc) * (1.f / 0.02f);   // sig = 2*0.1^2
    Kb[i] = expf(-0.5f * t * t);
}

__global__ __launch_bounds__(64)
void mi_hm_kernel(const float* __restrict__ Kb, float* __restrict__ Hm)
{
    __shared__ float sh[64];
    const int bc = blockIdx.x;              // b*96+c
    const int i  = threadIdx.x;
    float p = 0.f;
    for (int n = 0; n < 16; ++n) p += Kb[((size_t)(bc * 16 + n)) * 64 + i];
    p *= (1.f / 16.f);
    sh[i] = p; __syncthreads();
    for (int s = 32; s > 0; s >>= 1) { if (i < s) sh[i] += sh[i + s]; __syncthreads(); }
    const float psum = sh[0];
    __syncthreads();
    const float pn = p / (psum + 1e-10f);
    sh[i] = pn * log2f(pn + 1e-10f);
    __syncthreads();
    for (int s = 32; s > 0; s >>= 1) { if (i < s) sh[i] += sh[i + s]; __syncthreads(); }
    if (i == 0) Hm[bc] = -sh[0];
}

// Joint entropy per (b,s,d): J = Ks^T * Kd is a 64x64x16 GEMM -> 16 WMMA tiles
// (K zero-padded to 32). Sum/entropy are elementwise, so the D-tile lane
// mapping never needs decoding.
__global__ __launch_bounds__(128)
void mi_j_kernel(const float* __restrict__ Kb, const float* __restrict__ Hm,
                 float* __restrict__ miAcc)
{
    __shared__ __attribute__((aligned(32))) _Float16 KsT[64][32];  // (i, n) pad n>=16 = 0
    __shared__ __attribute__((aligned(32))) _Float16 KdT[64][32];  // (j, n) pad n>=16 = 0
    __shared__ float red[128];

    const int blk = blockIdx.x;             // b*2048 + s*64 + d
    const int d = blk & 63;
    const int s = (blk >> 6) & 31;
    const int b = blk >> 11;
    const int tid    = threadIdx.x;
    const int lane   = tid & 31;
    const int wave   = tid >> 5;
    const int half   = lane >> 4;
    const int lane16 = lane & 15;

    for (int e = tid; e < 1024; e += 128) {
        const int i = e >> 4, n = e & 15;
        KsT[i][n] = (_Float16)Kb[(((size_t)(b * 96 + s)) * 16 + n) * 64 + i];
        KdT[i][n] = (_Float16)Kb[(((size_t)(b * 96 + 32 + d)) * 16 + n) * 64 + i];
        KsT[i][16 + n] = (_Float16)0.f;
        KdT[i][16 + n] = (_Float16)0.f;
    }
    __syncthreads();

    // Wave w owns i-tile w; loops over 4 j-tiles.
    const v8h a0 = *(const v8h*)&KsT[wave * 16 + lane16][half * 8];
    const v8h a1 = *(const v8h*)&KsT[wave * 16 + lane16][16 + half * 8];
    const v16h af = __builtin_shufflevector(a0, a1, 0,1,2,3,4,5,6,7,8,9,10,11,12,13,14,15);

    v8f acc[4];
    float jsum = 0.f;
#pragma unroll
    for (int tj = 0; tj < 4; ++tj) {
        const v8h b0 = *(const v8h*)&KdT[tj * 16 + lane16][half * 8];
        const v8h b1 = *(const v8h*)&KdT[tj * 16 + lane16][16 + half * 8];
        const v16h bf = __builtin_shufflevector(b0, b1, 0,1,2,3,4,5,6,7,8,9,10,11,12,13,14,15);
        v8f z = {};
        acc[tj] = __builtin_amdgcn_wmma_f32_16x16x32_f16(false, af, false, bf,
                                                         (short)0, z, false, false);
#pragma unroll
        for (int r = 0; r < 8; ++r) jsum += acc[tj][r];
    }

    red[tid] = jsum; __syncthreads();
    for (int w = 64; w > 0; w >>= 1) { if (tid < w) red[tid] += red[tid + w]; __syncthreads(); }
    const float Jtot = red[0] + 1e-10f;
    __syncthreads();

    float ent = 0.f;
#pragma unroll
    for (int tj = 0; tj < 4; ++tj)
#pragma unroll
        for (int r = 0; r < 8; ++r) {
            const float jn = acc[tj][r] / Jtot;
            ent += jn * log2f(jn + 1e-10f);
        }
    red[tid] = ent; __syncthreads();
    for (int w = 64; w > 0; w >>= 1) { if (tid < w) red[tid] += red[tid + w]; __syncthreads(); }
    if (tid == 0) {
        const float Hj = -red[0];
        const float Hs = Hm[b * 96 + s];
        const float Hd = Hm[b * 96 + 32 + d];
        const float nmi = 2.f * (Hs + Hd - Hj) / (Hs + Hd);
        atomicAdd(miAcc, nmi * (1.f / 16384.f));
    }
}

// ---------------------------------------------------------------------------
// FC heads: mu/logvar/z, then decoder FC (relu)
// ---------------------------------------------------------------------------
__global__ void fc_z_kernel(const float* __restrict__ h5,
                            const float* __restrict__ Ws_mu, const float* __restrict__ bs_mu,
                            const float* __restrict__ Ws_lv, const float* __restrict__ bs_lv,
                            const float* __restrict__ Wd_mu, const float* __restrict__ bd_mu,
                            const float* __restrict__ Wd_lv, const float* __restrict__ bd_lv,
                            const float* __restrict__ eps,
                            float* __restrict__ muOut, float* __restrict__ lvOut,
                            float* __restrict__ zOut, float* __restrict__ zWs)
{
    const int i = blockIdx.x * blockDim.x + threadIdx.x;
    if (i >= 8 * 384) return;
    const int b = i / 384, j = i % 384;
    const float* hb = h5 + (size_t)b * 1536;
    float mu, lv;
    if (j < 128) {
        float m = bs_mu[j], l = bs_lv[j];
        for (int k = 0; k < 512; ++k) {
            const float hv = hb[k];
            m += hv * Ws_mu[k * 128 + j];
            l += hv * Ws_lv[k * 128 + j];
        }
        mu = m; lv = l;
    } else {
        const int jj = j - 128;
        const float* hd = hb + 512;
        float m = bd_mu[jj], l = bd_lv[jj];
        for (int k = 0; k < 1024; ++k) {
            const float hv = hd[k];
            m += hv * Wd_mu[k * 256 + jj];
            l += hv * Wd_lv[k * 256 + jj];
        }
        mu = m; lv = l;
    }
    const float z = eps[i] * expf(lv) + mu;
    muOut[i] = mu; lvOut[i] = lv; zOut[i] = z; zWs[i] = z;
}

__global__ void dec_fc_kernel(const float* __restrict__ z, const float* __restrict__ W,
                              const float* __restrict__ bias, float* __restrict__ g)
{
    const int i = blockIdx.x * blockDim.x + threadIdx.x;
    if (i >= 8 * 1536) return;
    const int b = i / 1536, j = i % 1536;
    float acc = bias[j];
    const float* zb = z + b * 384;
    for (int k = 0; k < 384; ++k) acc += zb[k] * W[k * 1536 + j];
    g[i] = fmaxf(acc, 0.f);
}

__global__ void init_scalar_kernel(float* p) { *p = 0.f; }

// ---------------------------------------------------------------------------
extern "C" void kernel_launch(void* const* d_in, const int* in_sizes, int n_in,
                              void* d_out, int out_size, void* d_ws, size_t ws_size,
                              hipStream_t stream)
{
    const float* x      = (const float*)d_in[0];
    const float* eps    = (const float*)d_in[1];
    const float* e_w1   = (const float*)d_in[2];
    const float* e_w2   = (const float*)d_in[3];
    const float* e_w3   = (const float*)d_in[4];
    const float* e_w4   = (const float*)d_in[5];
    const float* e_w5   = (const float*)d_in[6];
    const float* Ws_mu  = (const float*)d_in[7];
    const float* bs_mu  = (const float*)d_in[8];
    const float* Ws_lv  = (const float*)d_in[9];
    const float* bs_lv  = (const float*)d_in[10];
    const float* Wd_mu  = (const float*)d_in[11];
    const float* bd_mu  = (const float*)d_in[12];
    const float* Wd_lv  = (const float*)d_in[13];
    const float* bd_lv  = (const float*)d_in[14];
    const float* W_dec  = (const float*)d_in[15];
    const float* b_dec  = (const float*)d_in[16];
    const float* d_w1   = (const float*)d_in[17];
    const float* d_w2   = (const float*)d_in[18];
    const float* d_w3   = (const float*)d_in[19];
    const float* d_w4   = (const float*)d_in[20];
    const float* d_w5   = (const float*)d_in[21];

    float* out = (float*)d_out;
    float* ws  = (float*)d_ws;

    // Workspace layout (floats); encoder buffers reused by mirror decoder levels.
    float* A    = ws + 0;        // 2,097,152 : h1 (8,64,64,64)   -> dec4 out
    float* Bf   = ws + 2097152;  // 1,048,576 : h2 (8,128,32,32)  -> dec3 out
    float* C    = ws + 3145728;  //   524,288 : h3 (8,256,16,16)  -> dec2 out
    float* D    = ws + 3670016;  //   262,144 : h4 (8,512,8,8)    -> dec1 out
    float* E    = ws + 3932160;  //    12,288 : h5 (8,96,4,4)
    float* Kb   = ws + 3944448;  //   786,432 : KDE kernel matrix
    float* Hm   = ws + 4730880;  //       768 : marginal entropies
    float* mean = ws + 4731648;  //       512
    float* rstd = ws + 4732160;  //       512
    float* zWs  = ws + 4732672;  //     3,072
    float* g0   = ws + 4735744;  //    12,288 : decoder FC output (8,96,4,4)
    _Float16* wf   = (_Float16*)(ws + 4748032); // packed f16 weights
    _Float16* wf1  = wf;                        //      4,096 (64 x 64)
    _Float16* wf2  = wf + 4096;                 //    131,072 (128 x 1024)
    _Float16* wf3  = wf + 135168;               //    524,288 (256 x 2048)
    _Float16* wf4  = wf + 659456;               //  2,097,152 (512 x 4096)
    _Float16* wf5  = wf + 2756608;              //    786,432 (96 x 8192)
    _Float16* wfd1 = wf + 3543040;              //    786,432 (512 x 1536)
    _Float16* wfd2 = wf + 4329472;              //  2,097,152 (256 x 8192)
    _Float16* wfd3 = wf + 6426624;              //    524,288 (128 x 4096)
    _Float16* wfd4 = wf + 6950912;              //    131,072 (64 x 2048)
    _Float16* wfd5 = wf + 7081984;              //     16,384 (16 x 1024, M-padded)

    float* resOut = out;             // 393,216
    float* zOut   = out + 393216;    //   3,072
    float* muOut  = out + 396288;    //   3,072
    float* lvOut  = out + 399360;    //   3,072
    float* miOut  = out + 402432;    //       1

    init_scalar_kernel<<<1, 1, 0, stream>>>(miOut);

    // ---- pack weights to f16 ----
    pack_w_f16_kernel<<<(4096    + 255) / 256, 256, 0, stream>>>(e_w1, wf1,  64,   48,   64);
    pack_w_f16_kernel<<<(131072  + 255) / 256, 256, 0, stream>>>(e_w2, wf2, 128, 1024, 1024);
    pack_w_f16_kernel<<<(524288  + 255) / 256, 256, 0, stream>>>(e_w3, wf3, 256, 2048, 2048);
    pack_w_f16_kernel<<<(2097152 + 255) / 256, 256, 0, stream>>>(e_w4, wf4, 512, 4096, 4096);
    pack_w_f16_kernel<<<(786432  + 255) / 256, 256, 0, stream>>>(e_w5, wf5,  96, 8192, 8192);
    pack_wT_f16_kernel<<<(786432  + 255) / 256, 256, 0, stream>>>(d_w1, wfd1,  96, 512, 512);
    pack_wT_f16_kernel<<<(2097152 + 255) / 256, 256, 0, stream>>>(d_w2, wfd2, 512, 256, 256);
    pack_wT_f16_kernel<<<(524288  + 255) / 256, 256, 0, stream>>>(d_w3, wfd3, 256, 128, 128);
    pack_wT_f16_kernel<<<(131072  + 255) / 256, 256, 0, stream>>>(d_w4, wfd4, 128,  64,  64);
    pack_wT_f16_kernel<<<(16384   + 255) / 256, 256, 0, stream>>>(d_w5, wfd5,  64,   3,  16);

    // ---- encoder (WMMA implicit GEMM, TDM A-tiles, double-buffered) ----
    //                      x    w     out  Cin Cout Mrow IH   IW  ihw ow ohw  Kpad  md ep
    gemm_conv_wmma_kernel<<<dim3(4, 512), 128, 0, stream>>>(x,  wf1, A,    3,  64,  64, 128, 128, 14, 6, 12,   64, 0, 1);

    gemm_conv_wmma_kernel<<<dim3(8, 128), 128, 0, stream>>>(A,  wf2, Bf,  64, 128, 128,  64,  64, 12, 5, 10, 1024, 0, 0);
    bn_stats_kernel<<<128, 256, 0, stream>>>(Bf, mean, rstd, 8, 128, 10);
    bn_lrelu_kernel<<<1024, 256, 0, stream>>>(Bf, mean, rstd, 128, 10);

    gemm_conv_wmma_kernel<<<dim3(16, 32), 128, 0, stream>>>(Bf, wf3, C,  128, 256, 256,  32,  32, 10, 4,  8, 2048, 0, 0);
    bn_stats_kernel<<<256, 256, 0, stream>>>(C, mean, rstd, 8, 256, 8);
    bn_lrelu_kernel<<<2048, 256, 0, stream>>>(C, mean, rstd, 256, 8);

    gemm_conv_wmma_kernel<<<dim3(32, 8), 128, 0, stream>>>(C,  wf4, D,  256, 512, 512,  16,  16,  8, 3,  6, 4096, 0, 0);
    bn_stats_kernel<<<512, 256, 0, stream>>>(D, mean, rstd, 8, 512, 6);
    bn_lrelu_kernel<<<4096, 256, 0, stream>>>(D, mean, rstd, 512, 6);

    gemm_conv_wmma_kernel<<<dim3(6, 2), 128, 0, stream>>>(D,  wf5, E,  512,  96,  96,   8,   8,  6, 2,  4, 8192, 0, 0);
    bn_stats_kernel<<<96, 256, 0, stream>>>(E, mean, rstd, 8, 96, 4);
    bn_lrelu_kernel<<<768, 256, 0, stream>>>(E, mean, rstd, 96, 4);

    // ---- mutual information ----
    mi_k_kernel<<<786432 / 256, 256, 0, stream>>>(E, Kb);
    mi_hm_kernel<<<768, 64, 0, stream>>>(Kb, Hm);
    mi_j_kernel<<<16384, 128, 0, stream>>>(Kb, Hm, miOut);

    // ---- FC heads + reparameterization + decoder FC ----
    fc_z_kernel<<<(3072 + 255) / 256, 256, 0, stream>>>(E, Ws_mu, bs_mu, Ws_lv, bs_lv,
                                                        Wd_mu, bd_mu, Wd_lv, bd_lv, eps,
                                                        muOut, lvOut, zOut, zWs);
    dec_fc_kernel<<<(12288 + 255) / 256, 256, 0, stream>>>(zWs, W_dec, b_dec, g0);

    // ---- decoder (WMMA implicit GEMM convT) ----
    gemm_conv_wmma_kernel<<<dim3(32, 8), 128, 0, stream>>>(g0, wfd1, D,  96, 512, 512,  4,  4,  4, 3,  6, 1536, 1, 0);
    bn_stats_kernel<<<512, 256, 0, stream>>>(D, mean, rstd, 8, 512, 6);
    bn_lrelu_kernel<<<4096, 256, 0, stream>>>(D, mean, rstd, 512, 6);

    gemm_conv_wmma_kernel<<<dim3(16, 32), 128, 0, stream>>>(D, wfd2, C, 512, 256, 256,  8,  8,  6, 4,  8, 8192, 1, 0);
    bn_stats_kernel<<<256, 256, 0, stream>>>(C, mean, rstd, 8, 256, 8);
    bn_lrelu_kernel<<<2048, 256, 0, stream>>>(C, mean, rstd, 256, 8);

    gemm_conv_wmma_kernel<<<dim3(8, 128), 128, 0, stream>>>(C, wfd3, Bf, 256, 128, 128, 16, 16,  8, 5, 10, 4096, 1, 0);
    bn_stats_kernel<<<128, 256, 0, stream>>>(Bf, mean, rstd, 8, 128, 10);
    bn_lrelu_kernel<<<1024, 256, 0, stream>>>(Bf, mean, rstd, 128, 10);

    gemm_conv_wmma_kernel<<<dim3(4, 512), 128, 0, stream>>>(Bf, wfd4, A, 128,  64,  64, 32, 32, 10, 6, 12, 2048, 1, 0);
    bn_stats_kernel<<<64, 256, 0, stream>>>(A, mean, rstd, 8, 64, 12);
    bn_lrelu_kernel<<<512, 256, 0, stream>>>(A, mean, rstd, 64, 12);

    gemm_conv_wmma_kernel<<<dim3(1, 2048), 128, 0, stream>>>(A, wfd5, resOut, 64, 3, 16, 64, 64, 12, 7, 14, 1024, 1, 2);
}
